// LocalRVNet_61907658604935
// MI455X (gfx1250) — matrix operand order repaired and checked by reading, not compiled, not run on hardware
//
#include <hip/hip_runtime.h>

// ---------------------------------------------------------------------------
// LocalRVNet forward for MI455X (gfx1250, wave32, WMMA).
//
//   1) knn_kernel      : measure_coords (64KB) -> LDS via async DMA
//                        (global_load_async_to_lds_b128 / s_wait_asynccnt),
//                        per-thread register top-16 sorted insert, emit
//                        knn_feat [B,66] (d_out) + zero-padded X0 [B,96] (ws).
//   2) 5x gemm_bn      : Y = relu_bn(X)@W + b with v_wmma_f32_16x16x32_f16.
//                        Wave tile 16x64 (4 WMMAs share one A fragment);
//                        block tile 128x64. B staged N-major in LDS (2x
//                        ds_load_b128 per fragment), software-pipelined:
//                        next tile's global loads issue before the WMMAs.
//                        Input BN fused into A-load; output BN stats via
//                        ds_add_f32 + global atomics in the epilogue.
//      5x bn_finalize  : scale = g*rsqrt(var+eps), shift = beta - mean*scale.
//   3) final_layer     : fused BN5 + 64->2 GEMV (too narrow for WMMA).
// ---------------------------------------------------------------------------

typedef __attribute__((ext_vector_type(16))) _Float16 v16h;
typedef __attribute__((ext_vector_type(8)))  _Float16 v8h;
typedef __attribute__((ext_vector_type(2)))  _Float16 v2h;
typedef __attribute__((ext_vector_type(8)))  float    v8f;

#define BATCH 32768
#define MPTS  8192
#define KNN   16
#define D_IN  66
#define D_PAD 96
#define EPS   1e-5f

#define NT      4                 // n-tiles per wave (16x64 wave tile)
#define BN_COLS (NT * 16)         // 64 columns per block
#define LB_STR  40                // LDS B row stride in halves (80B, 16B-aligned)

// ---------------------------------------------------------------------------
// Phase 1: KNN + feature build
// ---------------------------------------------------------------------------
__global__ __launch_bounds__(256)
void knn_kernel(const float* __restrict__ pos,
                const float* __restrict__ mc,
                const float* __restrict__ mv,
                float* __restrict__ feat_out,   // [B,66] -> d_out tail
                float* __restrict__ x0)         // [B,96] padded, workspace
{
    __shared__ float2 smc[MPTS];                // 64 KB of 320 KB/WGP

    // Bulk-stage 64KB global -> LDS on the async path (ASYNCcnt, no VGPR data).
    {
        const unsigned ldsBase = (unsigned)(uintptr_t)(&smc[0]);
        const unsigned long long gBase = (unsigned long long)(uintptr_t)mc;
        for (int i = threadIdx.x; i < MPTS / 2; i += 256) {   // 16B chunks
            unsigned           ldsOff = ldsBase + (unsigned)i * 16u;
            unsigned long long gAddr  = gBase + (unsigned long long)i * 16ull;
            asm volatile("global_load_async_to_lds_b128 %0, %1, off"
                         :: "v"(ldsOff), "v"(gAddr) : "memory");
        }
        asm volatile("s_wait_asynccnt 0" ::: "memory");
    }
    __syncthreads();

    const int row = blockIdx.x * 256 + threadIdx.x;
    const float2 p = reinterpret_cast<const float2*>(pos)[row];

    float bestd[KNN];
    int   besti[KNN];
#pragma unroll
    for (int j = 0; j < KNN; ++j) { bestd[j] = 3.4e38f; besti[j] = 0; }

    for (int m = 0; m < MPTS; ++m) {
        float2 c = smc[m];
        float dx = p.x - c.x, dy = p.y - c.y;
        float d = dx * dx + dy * dy;
        if (d < bestd[KNN - 1]) {               // rare after warm-up
            float cd = d; int ci = m;
#pragma unroll
            for (int j = 0; j < KNN; ++j) {     // bubble-through sorted insert
                if (cd < bestd[j]) {
                    float td = bestd[j]; int ti = besti[j];
                    bestd[j] = cd; besti[j] = ci;
                    cd = td; ci = ti;
                }
            }
        }
    }

    float* f = feat_out + (size_t)row * D_IN;
    float* x = x0       + (size_t)row * D_PAD;
    f[0] = p.x; f[1] = p.y;
    x[0] = p.x; x[1] = p.y;
    const float2* mv2 = reinterpret_cast<const float2*>(mv);
#pragma unroll
    for (int j = 0; j < KNN; ++j) {
        float2 c = smc[besti[j]];
        float2 v = mv2[besti[j]];
        int o = 2 + 4 * j;
        f[o] = c.x; f[o + 1] = c.y; f[o + 2] = v.x; f[o + 3] = v.y;
        x[o] = c.x; x[o + 1] = c.y; x[o + 2] = v.x; x[o + 3] = v.y;
    }
#pragma unroll
    for (int t = D_IN; t < D_PAD; ++t) x[t] = 0.0f;
}

// ---------------------------------------------------------------------------
// Small helpers
// ---------------------------------------------------------------------------
__global__ void zero_kernel(float* p, int n)
{
    int i = blockIdx.x * blockDim.x + threadIdx.x;
    if (i < n) p[i] = 0.0f;
}

__global__ void pad_w1_kernel(const float* __restrict__ w1, float* __restrict__ w1p)
{
    int i = blockIdx.x * blockDim.x + threadIdx.x;   // over 96*128
    if (i < D_PAD * 128) {
        int k = i / 128, n = i % 128;
        w1p[i] = (k < D_IN) ? w1[k * 128 + n] : 0.0f;
    }
}

// ---------------------------------------------------------------------------
// Phase 2: WMMA GEMM + fused input-BN + output-BN statistics.
// Block = 256 threads (8 waves). Block tile: 128 rows x 64 cols.
// Wave w owns rows [blockIdx.y*128 + 16w, +16) and all 64 cols (4 WMMA tiles).
// gridDim.x = Nout/64, gridDim.y = BATCH/128.
// ---------------------------------------------------------------------------
__global__ __launch_bounds__(256)
void gemm_bn_kernel(const float* __restrict__ X,
                    const float* __restrict__ W,
                    const float* __restrict__ bias,
                    float* __restrict__ Y,
                    float* __restrict__ gsum,
                    float* __restrict__ gsq,
                    const float* __restrict__ inSc,   // BN scale over Kin (may be null)
                    const float* __restrict__ inSh,   // BN shift over Kin
                    int Kin, int Nout, int applyIn)
{
    __shared__ _Float16 lB[BN_COLS][LB_STR];   // N-major staged B tile
    __shared__ float s_sum[BN_COLS], s_sq[BN_COLS];

    const int tid  = threadIdx.x;
    const int lane = tid & 31;
    const int wave = tid >> 5;
    const int n0   = blockIdx.x * BN_COLS;
    const int rows = blockIdx.y * 128 + wave * 16;

    if (tid < BN_COLS) { s_sum[tid] = 0.0f; s_sq[tid] = 0.0f; }

    v8f c[NT];
#pragma unroll
    for (int t = 0; t < NT; ++t) c[t] = (v8f){0.f,0.f,0.f,0.f,0.f,0.f,0.f,0.f};

    const int row  = rows + (lane & 15);
    const int koff = (lane & 16) ? 8 : 0;     // ISA A-layout: hi lanes take K+8
    const int ks   = (lane & 16) ? 8 : 0;     // ISA B-layout: hi lanes take K+8
    const int nn   = lane & 15;

    // ---- B staging: fixed LDS addresses + striding integer offsets into W --
    // (integer offsets keep the loads provably global -> global_load_b32,
    //  not flat_load_b32 which would also burn DScnt / LDS issue slots)
    size_t woff[NT];
    v2h*   stAddr[NT];
#pragma unroll
    for (int j = 0; j < NT; ++j) {
        int p  = tid + 256 * j;               // pair index over 1024
        int kk = (p >> 6) << 1;               // even K row
        int nc = p & 63;                      // column
        woff[j]   = (size_t)kk * Nout + n0 + nc;
        stAddr[j] = (v2h*)&lB[nc][kk];
    }
    const size_t wStep = (size_t)32 * Nout;

    // prologue: load first B tile into registers
    float lo[NT], hi[NT];
#pragma unroll
    for (int j = 0; j < NT; ++j) {
        lo[j] = W[woff[j]];                   // K = kk
        hi[j] = W[woff[j] + Nout];            // K = kk+1
        woff[j] += wStep;
    }

    const float* xpBase = X + (size_t)row * Kin + koff;

    for (int k0 = 0; k0 < Kin; k0 += 32) {
        __syncthreads();                      // previous-iter LDS reads done
        // ---- store staged B tile (packed half2, straight-line) ----
#pragma unroll
        for (int j = 0; j < NT; ++j) {
            v2h pk;
            pk[0] = (_Float16)lo[j];
            pk[1] = (_Float16)hi[j];
            *stAddr[j] = pk;                  // ds_store_b32
        }
        __syncthreads();

        // ---- issue next tile's global loads early (hidden under WMMAs) ----
        if (k0 + 32 < Kin) {
#pragma unroll
            for (int j = 0; j < NT; ++j) {
                lo[j] = W[woff[j]];
                hi[j] = W[woff[j] + Nout];
                woff[j] += wStep;
            }
        }

        // ---- A fragment: rows of X, fused relu(scale*x + shift) ----
        const float* xp = xpBase + k0;
        float fa[8], fb[8];
#pragma unroll
        for (int i = 0; i < 8; ++i) { fa[i] = xp[i]; fb[i] = xp[16 + i]; }
        if (applyIn) {
#pragma unroll
            for (int i = 0; i < 8; ++i) {
                int ka = k0 + koff + i, kb = ka + 16;
                fa[i] = fmaxf(0.f, fmaf(fa[i], inSc[ka], inSh[ka]));
                fb[i] = fmaxf(0.f, fmaf(fb[i], inSc[kb], inSh[kb]));
            }
        }
        v16h a;
#pragma unroll
        for (int v = 0; v < 4; ++v) {         // VGPR v: K = 2v, 2v+1 (+koff)
            a[2 * v]     = (_Float16)fa[2 * v];
            a[2 * v + 1] = (_Float16)fa[2 * v + 1];
        }
#pragma unroll
        for (int v = 4; v < 8; ++v) {         // VGPR v: K = 16+2(v-4) (+koff)
            a[2 * v]     = (_Float16)fb[2 * (v - 4)];
            a[2 * v + 1] = (_Float16)fb[2 * (v - 4) + 1];
        }

        // ---- 4 WMMAs reuse the A fragment; B frag = 2x ds_load_b128 ----
#pragma unroll
        for (int t = 0; t < NT; ++t) {
            const _Float16* bp = &lB[t * 16 + nn][ks];
            v8h b0 = *(const v8h*)(bp);
            v8h b1 = *(const v8h*)(bp + 16);
            v16h b = __builtin_shufflevector(b0, b1,
                        0,1,2,3,4,5,6,7,8,9,10,11,12,13,14,15);
            c[t] = __builtin_amdgcn_wmma_f32_16x16x32_f16(
                       false, a, false, b, (short)0, c[t], false, false);
        }
    }

    // ---- epilogue: bias, store, per-column sum/sumsq for BN ----
    const int rbase = rows + ((lane & 16) ? 8 : 0);   // C/D layout lane split
#pragma unroll
    for (int t = 0; t < NT; ++t) {
        const int n    = n0 + t * 16 + nn;
        const float bv = bias[n];
        float ps = 0.f, pq = 0.f;
#pragma unroll
        for (int r = 0; r < 8; ++r) {
            float y = c[t][r] + bv;
            Y[(size_t)(rbase + r) * Nout + n] = y;
            ps += y; pq += y * y;
        }
        atomicAdd(&s_sum[t * 16 + nn], ps);          // ds_add_f32
        atomicAdd(&s_sq[t * 16 + nn], pq);
    }
    __syncthreads();
    if (tid < BN_COLS) {
        atomicAdd(&gsum[n0 + tid], s_sum[tid]);
        atomicAdd(&gsq[n0 + tid],  s_sq[tid]);
    }
}

__global__ void bn_finalize_kernel(const float* __restrict__ gsum,
                                   const float* __restrict__ gsq,
                                   const float* __restrict__ g,
                                   const float* __restrict__ be,
                                   float* __restrict__ scale,
                                   float* __restrict__ shift,
                                   int N)
{
    int n = threadIdx.x;
    if (n < N) {
        float inv_b = 1.0f / (float)BATCH;
        float mean = gsum[n] * inv_b;
        float var  = gsq[n] * inv_b - mean * mean;   // biased, matches var(0)
        float s = g[n] * rsqrtf(var + EPS);
        scale[n] = s;
        shift[n] = be[n] - mean * s;
    }
}

// ---------------------------------------------------------------------------
// Phase 3: fused BN5 + 64->2 head (too narrow for a 16-wide WMMA tile).
// ---------------------------------------------------------------------------
__global__ __launch_bounds__(256)
void final_layer_kernel(const float* __restrict__ X5,
                        const float* __restrict__ w6,
                        const float* __restrict__ b6,
                        const float* __restrict__ sc,
                        const float* __restrict__ sh,
                        float* __restrict__ out)
{
    __shared__ float sw[128], ssc[64], ssh[64], sb[2];
    int tid = threadIdx.x;
    if (tid < 128) sw[tid] = w6[tid];
    if (tid < 64) { ssc[tid] = sc[tid]; ssh[tid] = sh[tid]; }
    if (tid < 2)   sb[tid] = b6[tid];
    __syncthreads();

    const int row = blockIdx.x * 256 + tid;
    const float* xr = X5 + (size_t)row * 64;
    float a0 = 0.f, a1 = 0.f;
#pragma unroll
    for (int k = 0; k < 64; ++k) {
        float x = fmaxf(0.f, fmaf(xr[k], ssc[k], ssh[k]));
        a0 = fmaf(x, sw[2 * k], a0);
        a1 = fmaf(x, sw[2 * k + 1], a1);
    }
    out[2 * row]     = a0 + sb[0];
    out[2 * row + 1] = a1 + sb[1];
}

// ---------------------------------------------------------------------------
// Launcher
// ---------------------------------------------------------------------------
extern "C" void kernel_launch(void* const* d_in, const int* in_sizes, int n_in,
                              void* d_out, int out_size, void* d_ws, size_t ws_size,
                              hipStream_t stream)
{
    const float* pos = (const float*)d_in[0];
    const float* mc  = (const float*)d_in[1];
    const float* mv  = (const float*)d_in[2];
    const float* w[7]  = {nullptr, (const float*)d_in[3], (const float*)d_in[5],
                          (const float*)d_in[7], (const float*)d_in[9],
                          (const float*)d_in[11], (const float*)d_in[13]};
    const float* bb[7] = {nullptr, (const float*)d_in[4], (const float*)d_in[6],
                          (const float*)d_in[8], (const float*)d_in[10],
                          (const float*)d_in[12], (const float*)d_in[14]};
    const float* gg[6]  = {nullptr, (const float*)d_in[15], (const float*)d_in[17],
                           (const float*)d_in[19], (const float*)d_in[21],
                           (const float*)d_in[23]};
    const float* bee[6] = {nullptr, (const float*)d_in[16], (const float*)d_in[18],
                           (const float*)d_in[20], (const float*)d_in[22],
                           (const float*)d_in[24]};

    float* out  = (float*)d_out;                     // [B,2] then [B,66]
    float* feat = out + (size_t)BATCH * 2;

    // workspace layout (floats): P | Q | W1p | stats(sum,sq) | bn(scale,shift)
    float* ws = (float*)d_ws;
    float* P      = ws;                              // 32768*256
    float* Q      = P + (size_t)BATCH * 256;         // 32768*512
    float* w1p    = Q + (size_t)BATCH * 512;         // 96*128
    float* stats  = w1p + (size_t)D_PAD * 128;       // 5 * (512+512)
    float* bnpar  = stats + 5 * 1024;                // 5 * (512+512)

    float* sum_[5], *sq_[5], *sc_[5], *sh_[5];
    for (int l = 0; l < 5; ++l) {
        sum_[l] = stats + l * 1024;  sq_[l] = sum_[l] + 512;
        sc_[l]  = bnpar + l * 1024;  sh_[l] = sc_[l] + 512;
    }

    zero_kernel<<<(5 * 1024 + 255) / 256, 256, 0, stream>>>(stats, 5 * 1024);
    pad_w1_kernel<<<(D_PAD * 128 + 255) / 256, 256, 0, stream>>>(w[1], w1p);
    knn_kernel<<<BATCH / 256, 256, 0, stream>>>(pos, mc, mv, feat, P);

    const int ROWB = BATCH / 128;                    // 256 row-blocks
    // L1: X0[B,96] -> Y1[B,128] (no input BN)
    gemm_bn_kernel<<<dim3(128 / BN_COLS, ROWB), 256, 0, stream>>>(
        P, w1p, bb[1], Q, sum_[0], sq_[0], nullptr, nullptr, D_PAD, 128, 0);
    bn_finalize_kernel<<<1, 128, 0, stream>>>(sum_[0], sq_[0], gg[1], bee[1], sc_[0], sh_[0], 128);
    // L2: 128 -> 256
    gemm_bn_kernel<<<dim3(256 / BN_COLS, ROWB), 256, 0, stream>>>(
        Q, w[2], bb[2], P, sum_[1], sq_[1], sc_[0], sh_[0], 128, 256, 1);
    bn_finalize_kernel<<<1, 256, 0, stream>>>(sum_[1], sq_[1], gg[2], bee[2], sc_[1], sh_[1], 256);
    // L3: 256 -> 512
    gemm_bn_kernel<<<dim3(512 / BN_COLS, ROWB), 256, 0, stream>>>(
        P, w[3], bb[3], Q, sum_[2], sq_[2], sc_[1], sh_[1], 256, 512, 1);
    bn_finalize_kernel<<<1, 512, 0, stream>>>(sum_[2], sq_[2], gg[3], bee[3], sc_[2], sh_[2], 512);
    // L4: 512 -> 128
    gemm_bn_kernel<<<dim3(128 / BN_COLS, ROWB), 256, 0, stream>>>(
        Q, w[4], bb[4], P, sum_[3], sq_[3], sc_[2], sh_[2], 512, 128, 1);
    bn_finalize_kernel<<<1, 128, 0, stream>>>(sum_[3], sq_[3], gg[4], bee[4], sc_[3], sh_[3], 128);
    // L5: 128 -> 64
    gemm_bn_kernel<<<dim3(64 / BN_COLS, ROWB), 256, 0, stream>>>(
        P, w[5], bb[5], Q, sum_[4], sq_[4], sc_[3], sh_[3], 128, 64, 1);
    bn_finalize_kernel<<<1, 64, 0, stream>>>(sum_[4], sq_[4], gg[5], bee[5], sc_[4], sh_[4], 64);
    // L6: fused BN5 + 64 -> 2 head
    final_layer_kernel<<<BATCH / 256, 256, 0, stream>>>(Q, w[6], bb[6], sc_[4], sh_[4], out);

    (void)in_sizes; (void)n_in; (void)out_size; (void)ws_size;
}
